// GAS_59622736003074
// MI455X (gfx1250) — matrix operand order repaired and checked by reading, not compiled, not run on hardware
//
#include <hip/hip_runtime.h>
#include <hip/hip_bf16.h>
#include <math.h>

#define CIN   256
#define COUT  128
#define NGRAPH 32
#define BN_EPS 1e-5f
#define NEG_SLOPE 0.2f
#define MAXN_PER_GRAPH 2048

typedef __attribute__((ext_vector_type(16))) __bf16 v16bf;
typedef __attribute__((ext_vector_type(8)))  float  v8f;

// ---------------------------------------------------------------- zero fill
__global__ void k_zero(float* __restrict__ p, size_t n) {
  size_t i = (size_t)blockIdx.x * blockDim.x + threadIdx.x;
  size_t stride = (size_t)gridDim.x * blockDim.x;
  for (; i < n; i += stride) p[i] = 0.0f;
}

// ---------------------------------------------------------------- GEMM: xw = x @ W  (bf16 WMMA, f32 accum)
// block = 256 threads (8 waves); block tile = 128 rows x 128 cols; K = 256.
__global__ __launch_bounds__(256) void k_gemm_xw(const float* __restrict__ x,
                                                 const float* __restrict__ W,
                                                 float* __restrict__ xw) {
  __shared__ __bf16 sW[COUT][CIN];   // transposed: sW[n][k], 64 KB
  const int tid = threadIdx.x;
  // stage W (row-major [k][n] in memory) transposed into LDS as bf16
  for (int i = tid; i < CIN * COUT; i += 256) {
    int kk = i / COUT;
    int nn = i % COUT;                 // consecutive threads -> consecutive n: coalesced global read
    sW[nn][kk] = (__bf16)W[kk * COUT + nn];
  }
  __syncthreads();

  const int wave = tid >> 5;
  const int lane = tid & 31;
  const int half = lane >> 4;          // 0: lanes 0-15, 1: lanes 16-31
  const int l16  = lane & 15;
  const int rowBase = blockIdx.x * 128 + wave * 16;

  v8f acc[8];
  #pragma unroll
  for (int cg = 0; cg < 8; ++cg)
    #pragma unroll
    for (int r = 0; r < 8; ++r) acc[cg][r] = 0.0f;

  // A: each lane holds row M = l16 (both halves), K split per ISA 16-bit A 16x32 layout
  const float* xrow = x + (size_t)(rowBase + l16) * CIN;

  for (int kc = 0; kc < CIN; kc += 32) {
    v16bf a;
    #pragma unroll
    for (int v = 0; v < 8; ++v) {
      int k = kc + ((v < 4) ? (8 * half + 2 * v) : (16 + 8 * half + 2 * (v - 4)));
      a[2 * v]     = (__bf16)xrow[k];
      a[2 * v + 1] = (__bf16)xrow[k + 1];
    }
    #pragma unroll
    for (int cg = 0; cg < 8; ++cg) {
      // B: 32x16 per ISA layout; lane l16 = column within group, K split same as A (8*half + s, 16 + 8*half + s-8)
      v16bf b;
      const int col = cg * 16 + l16;
      #pragma unroll
      for (int v = 0; v < 8; ++v) {
        int k = kc + ((v < 4) ? (8 * half + 2 * v) : (16 + 8 * half + 2 * (v - 4)));
        b[2 * v]     = sW[col][k];
        b[2 * v + 1] = sW[col][k + 1];
      }
      acc[cg] = __builtin_amdgcn_wmma_f32_16x16x32_bf16(
          false, a, false, b, (short)0, acc[cg], false, false);
    }
  }

  // D layout: VGPR r -> M = r + 8*half, N = lane%16
  #pragma unroll
  for (int cg = 0; cg < 8; ++cg)
    #pragma unroll
    for (int r = 0; r < 8; ++r) {
      int m = rowBase + r + 8 * half;
      int n = cg * 16 + l16;
      xw[(size_t)m * COUT + n] = acc[cg][r];
    }
}

// ---------------------------------------------------------------- per-node attention dots
__global__ void k_att(const float* __restrict__ xw, const float* __restrict__ att,
                      float* __restrict__ a_i, float* __restrict__ a_j, int N) {
  int i = blockIdx.x * blockDim.x + threadIdx.x;
  if (i >= N) return;
  const float* r = xw + (size_t)i * COUT;
  float si = 0.f, sj = 0.f;
  #pragma unroll 4
  for (int c = 0; c < COUT; ++c) {
    float v = r[c];
    si += v * att[c];
    sj += v * att[COUT + c];
  }
  a_i[i] = si;
  a_j[i] = sj;
}

// monotone float->uint encoding for atomic float max
__device__ __forceinline__ unsigned fenc(float f) {
  unsigned b = __float_as_uint(f);
  return (b & 0x80000000u) ? ~b : (b | 0x80000000u);
}
__device__ __forceinline__ float fdec(unsigned m) {
  unsigned b = (m & 0x80000000u) ? (m & 0x7fffffffu) : ~m;
  return __uint_as_float(b);
}

// ---------------------------------------------------------------- edge pass 1: raw alpha + leaky relu + seg max
__global__ void k_alpha1(const int* __restrict__ ei, const float* __restrict__ a_i,
                         const float* __restrict__ a_j, float* __restrict__ alpha,
                         unsigned* __restrict__ segmax, int E, int total) {
  int e = blockIdx.x * blockDim.x + threadIdx.x;
  if (e >= total) return;
  int src = (e < E) ? ei[e]     : (e - E);
  int dst = (e < E) ? ei[E + e] : (e - E);
  float al = a_i[dst] + a_j[src];
  al = (al >= 0.f) ? al : NEG_SLOPE * al;
  alpha[e] = al;
  atomicMax(&segmax[dst], fenc(al));
}

// ---------------------------------------------------------------- edge pass 2: exp + denom
__global__ void k_alpha2(const int* __restrict__ ei, float* __restrict__ alpha,
                         const unsigned* __restrict__ segmax, float* __restrict__ denom,
                         int E, int total) {
  int e = blockIdx.x * blockDim.x + threadIdx.x;
  if (e >= total) return;
  int dst = (e < E) ? ei[E + e] : (e - E);
  float smax = fdec(segmax[dst]);
  if (!isfinite(smax)) smax = 0.f;
  float ex = __expf(alpha[e] - smax);
  alpha[e] = ex;
  atomicAdd(&denom[dst], ex);
}

// ---------------------------------------------------------------- edge pass 3: normalize + alpha_sum by src
__global__ void k_alpha3(const int* __restrict__ ei, float* __restrict__ alpha,
                         const float* __restrict__ denom, float* __restrict__ alpha_sum,
                         int E, int total) {
  int e = blockIdx.x * blockDim.x + threadIdx.x;
  if (e >= total) return;
  int src = (e < E) ? ei[e]     : (e - E);
  int dst = (e < E) ? ei[E + e] : (e - E);
  float al = alpha[e] / denom[dst];
  alpha[e] = al;
  atomicAdd(&alpha_sum[src], al);
}

// ---------------------------------------------------------------- per-graph exact top-k via rank (stable ties)
__global__ __launch_bounds__(256) void k_topk(const float* __restrict__ alpha_sum,
                                              int* __restrict__ perm_ws,
                                              int* __restrict__ perm_out,
                                              int* __restrict__ batch_out,
                                              int* __restrict__ keep,
                                              int n, int k) {
  __shared__ float sS[MAXN_PER_GRAPH];
  const int g = blockIdx.x;
  const int base = g * n;
  for (int j = threadIdx.x; j < n; j += blockDim.x) sS[j] = alpha_sum[base + j];
  __syncthreads();
  for (int j = threadIdx.x; j < n; j += blockDim.x) {
    float sj = sS[j];
    int rank = 0;
    for (int j2 = 0; j2 < n; ++j2) {
      float s2 = sS[j2];
      rank += (s2 > sj) || (s2 == sj && j2 < j);
    }
    if (rank < k) {
      int node = base + j;
      perm_ws[g * k + rank]  = node;
      perm_out[g * k + rank] = node;
      batch_out[g * k + rank] = g;
      keep[node] = 1;
    }
  }
}

// ---------------------------------------------------------------- masked scatter-add aggregation (1 wave / edge)
__global__ __launch_bounds__(256) void k_aggr(const int* __restrict__ ei,
                                              const float* __restrict__ xw,
                                              const float* __restrict__ alpha,
                                              const int* __restrict__ keep,
                                              float* __restrict__ out, int E, int total) {
  int e = blockIdx.x * 8 + (threadIdx.x >> 5);
  int lane = threadIdx.x & 31;
  if (e >= total) return;
  int src = (e < E) ? ei[e]     : (e - E);
  int dst = (e < E) ? ei[E + e] : (e - E);
  if (!(keep[src] && keep[dst])) return;
  float sc = alpha[e];
  const float4* xr = (const float4*)(xw + (size_t)src * COUT);
  float4 v = xr[lane];
  float* o = out + (size_t)dst * COUT + lane * 4;
  atomicAdd(o + 0, v.x * sc);
  atomicAdd(o + 1, v.y * sc);
  atomicAdd(o + 2, v.z * sc);
  atomicAdd(o + 3, v.w * sc);
}

// ---------------------------------------------------------------- batch-norm statistics
__global__ __launch_bounds__(128) void k_bnstat(const float* __restrict__ out,
                                                const int* __restrict__ perm,
                                                float* __restrict__ sums,
                                                float* __restrict__ sumsq, int rows) {
  int c = threadIdx.x;
  int r0 = blockIdx.x * 256;
  float s = 0.f, q = 0.f;
  for (int r = r0; r < r0 + 256 && r < rows; ++r) {
    float v = out[(size_t)perm[r] * COUT + c];
    s += v;
    q += v * v;
  }
  atomicAdd(&sums[c], s);
  atomicAdd(&sumsq[c], q);
}

// ---------------------------------------------------------------- gather + batch-norm apply
__global__ void k_bnapply(const float* __restrict__ out, const int* __restrict__ perm,
                          const float* __restrict__ sums, const float* __restrict__ sumsq,
                          const float* __restrict__ gamma, const float* __restrict__ beta,
                          const int* __restrict__ bnflag, float* __restrict__ dout, int rows) {
  int idx = blockIdx.x * blockDim.x + threadIdx.x;
  if (idx >= rows * COUT) return;
  int r = idx / COUT, c = idx % COUT;
  float v = out[(size_t)perm[r] * COUT + c];
  if (*bnflag) {
    float inv = 1.0f / (float)rows;
    float mu  = sums[c] * inv;
    float var = sumsq[c] * inv - mu * mu;
    v = (v - mu) * rsqrtf(var + BN_EPS) * gamma[c] + beta[c];
  }
  dout[idx] = v;
}

// ================================================================ launch
extern "C" void kernel_launch(void* const* d_in, const int* in_sizes, int n_in,
                              void* d_out, int out_size, void* d_ws, size_t ws_size,
                              hipStream_t stream) {
  const float* x        = (const float*)d_in[0];
  const float* W        = (const float*)d_in[1];
  const float* att      = (const float*)d_in[2];
  const float* bn_gamma = (const float*)d_in[3];
  const float* bn_beta  = (const float*)d_in[4];
  const int*   ei       = (const int*)d_in[5];
  const int*   bnflag   = (const int*)d_in[7];

  const int N     = in_sizes[0] / CIN;      // 65536
  const int E     = in_sizes[5] / 2;        // 1048576
  const int total = E + N;                  // with self loops
  const int n     = N / NGRAPH;             // 2048 nodes per graph
  const int k     = n / 2;                  // 1024 kept per graph
  const int rows  = NGRAPH * k;             // 32768 output rows

  // workspace layout (all f32-sized slots)
  float*    xw        = (float*)d_ws;                         // N*COUT
  float*    a_i       = xw + (size_t)N * COUT;                // N
  float*    a_j       = a_i + N;                              // N
  float*    alpha     = a_j + N;                              // total
  // ---- contiguous zero span starts here ----
  unsigned* segmax    = (unsigned*)(alpha + total);           // N
  float*    denom     = (float*)(segmax + N);                 // N
  float*    alpha_sum = denom + N;                            // N
  int*      keep      = (int*)(alpha_sum + N);                // N
  float*    sums      = (float*)(keep + N);                   // COUT
  float*    sumsq     = sums + COUT;                          // COUT
  float*    outbuf    = sumsq + COUT;                         // N*COUT
  // ---- zero span ends ----
  int*      perm_ws   = (int*)(outbuf + (size_t)N * COUT);    // rows

  // output layout: [out_final rows*COUT f32][perm rows i32][topk_batch rows i32]
  float* out_final = (float*)d_out;
  int*   perm_out  = (int*)(out_final + (size_t)rows * COUT);
  int*   batch_out = perm_out + rows;

  const size_t zero_elems = (size_t)4 * N + 2 * COUT + (size_t)N * COUT;
  k_zero<<<4096, 256, 0, stream>>>((float*)segmax, zero_elems);

  k_gemm_xw<<<N / 128, 256, 0, stream>>>(x, W, xw);

  k_att<<<(N + 255) / 256, 256, 0, stream>>>(xw, att, a_i, a_j, N);

  const int egrid = (total + 255) / 256;
  k_alpha1<<<egrid, 256, 0, stream>>>(ei, a_i, a_j, alpha, segmax, E, total);
  k_alpha2<<<egrid, 256, 0, stream>>>(ei, alpha, segmax, denom, E, total);
  k_alpha3<<<egrid, 256, 0, stream>>>(ei, alpha, denom, alpha_sum, E, total);

  k_topk<<<NGRAPH, 256, 0, stream>>>(alpha_sum, perm_ws, perm_out, batch_out, keep, n, k);

  k_aggr<<<(total + 7) / 8, 256, 0, stream>>>(ei, xw, alpha, keep, outbuf, E, total);

  k_bnstat<<<(rows + 255) / 256, 128, 0, stream>>>(outbuf, perm_ws, sums, sumsq, rows);

  k_bnapply<<<((size_t)rows * COUT + 255) / 256, 256, 0, stream>>>(
      outbuf, perm_ws, sums, sumsq, bn_gamma, bn_beta, bnflag, out_final, rows);
}
